// SelfAttentionHead_33440615367344
// MI455X (gfx1250) — matrix operand order repaired and checked by reading, not compiled, not run on hardware
//
#include <hip/hip_runtime.h>
#include <hip/hip_bf16.h>

// ---------------------------------------------------------------------------
// SelfAttentionHead (B=8, T=4096, DM=1024, DK=DV=128, NFIX=64) for gfx1250.
// bf16 WMMA (v_wmma_f32_16x16x32_bf16) for all four GEMMs; flash attention
// with online softmax; V stored transposed so PV B-fragments are contiguous;
// P tile staged through LDS for the C/D -> A layout change.
// ---------------------------------------------------------------------------

typedef __attribute__((ext_vector_type(16))) __bf16       v16bf;
typedef __attribute__((ext_vector_type(8)))  float        v8f;
typedef __attribute__((ext_vector_type(4)))  unsigned int u32x4;
typedef __attribute__((ext_vector_type(4)))  float        f32x4;

#define DM   1024
#define DK   128
#define DV   128
#define NFIX 64
#define BB   8
#define TT   4096
#define SKV  (TT + NFIX)      // 4160 keys/values per batch (fixed slots first)
#define MTOT (BB * TT)        // 32768 tokens

union V16U { v16bf v; u32x4 q[2]; };

__device__ __forceinline__ v16bf load_bf16x16(const unsigned short* p) {
  V16U t;
  t.q[0] = *(const u32x4*)(p);
  t.q[1] = *(const u32x4*)(p + 8);
  return t.v;
}

__device__ __forceinline__ unsigned short f2bf(float f) {
  __bf16 h = (__bf16)f;
  return __builtin_bit_cast(unsigned short, h);
}

// ---------------------------------------------------------------------------
// Kernel 0: fp32 weights -> bf16 workspace [3][128][1024]
// ---------------------------------------------------------------------------
__global__ __launch_bounds__(256) void prep_weights(
    const float* __restrict__ wq, const float* __restrict__ wk,
    const float* __restrict__ wv, unsigned short* __restrict__ wb) {
  int i = blockIdx.x * 256 + threadIdx.x;          // 0 .. 3*128*1024-1
  const int per = DK * DM;                          // 131072
  const float* src = (i < per) ? wq : ((i < 2 * per) ? wk : wv);
  wb[i] = f2bf(src[i % per]);
}

// ---------------------------------------------------------------------------
// Kernel 1: fixed slots -> Kx[b][0..63][:] (bf16), Vt[b][:][0..63]
//           (bf16, transposed layout [B][DV][SKV]).
// ---------------------------------------------------------------------------
__global__ __launch_bounds__(256) void prep_fixed(
    const float* __restrict__ fk, const float* __restrict__ fv,
    unsigned short* __restrict__ Kx, unsigned short* __restrict__ Vt) {
  int i = blockIdx.x * 256 + threadIdx.x;          // 0 .. 8*64*128-1
  int b   = i >> 13;
  int rem = i & 8191;
  int s   = rem >> 7;
  int d   = rem & 127;
  Kx[((size_t)b * SKV + s) * DK + d] = f2bf(fk[s * DK + d]);
  Vt[((size_t)b * DV + d) * SKV + s] = f2bf(fv[s * DV + d]);
}

// ---------------------------------------------------------------------------
// Kernel 2: QKV projection. One wave computes a 32-token x 128-feature tile
// for one of {Q,K,V}: 2 A fragments share each B fragment -> 16 WMMAs per
// 24 b128 loads per K-chunk (round 1 was 8 per 20).
// Layouts (per CDNA5 ISA tables):
//   A (16x32 bf16): lane row = lane&15; K halves {kb..kb+7},{16+kb..} with
//                   kb = (lane>>4)*8.
//   B (32x16 bf16): same K halves as A ({kb..kb+7},{16+kb..}); col=lane&15.
//   C/D (16x16 f32): VGPR i holds row (lane>>4)*8+i, col lane&15.
// ---------------------------------------------------------------------------
__global__ __launch_bounds__(32) void qkv_proj(
    const float* __restrict__ x, const unsigned short* __restrict__ Wb,
    unsigned short* __restrict__ Q, unsigned short* __restrict__ Kx,
    unsigned short* __restrict__ Vt) {
  const int lane = threadIdx.x;
  const int r    = lane & 15;
  const int half = lane >> 4;
  const int kb   = half * 8;
  const int m0   = blockIdx.x * 32;                 // token tile base (32 rows)
  const int mat  = blockIdx.y;                      // 0=Q 1=K 2=V

  const float*          xrow0 = x + (size_t)(m0 + r) * DM;
  const float*          xrow1 = x + (size_t)(m0 + 16 + r) * DM;
  const unsigned short* w     = Wb + (size_t)mat * DK * DM;

  v8f acc0[8] = {};
  v8f acc1[8] = {};

  for (int c0 = 0; c0 < DM; c0 += 32) {
    // ---- two A fragments (fp32 -> bf16 in-register) ----
    v16bf a0, a1;
    {
      f32x4 t0 = *(const f32x4*)(xrow0 + c0 + kb);
      f32x4 t1 = *(const f32x4*)(xrow0 + c0 + kb + 4);
      f32x4 t2 = *(const f32x4*)(xrow0 + c0 + kb + 16);
      f32x4 t3 = *(const f32x4*)(xrow0 + c0 + kb + 20);
      a0[0]=(__bf16)t0[0]; a0[1]=(__bf16)t0[1]; a0[2]=(__bf16)t0[2]; a0[3]=(__bf16)t0[3];
      a0[4]=(__bf16)t1[0]; a0[5]=(__bf16)t1[1]; a0[6]=(__bf16)t1[2]; a0[7]=(__bf16)t1[3];
      a0[8]=(__bf16)t2[0]; a0[9]=(__bf16)t2[1]; a0[10]=(__bf16)t2[2]; a0[11]=(__bf16)t2[3];
      a0[12]=(__bf16)t3[0]; a0[13]=(__bf16)t3[1]; a0[14]=(__bf16)t3[2]; a0[15]=(__bf16)t3[3];
    }
    {
      f32x4 t0 = *(const f32x4*)(xrow1 + c0 + kb);
      f32x4 t1 = *(const f32x4*)(xrow1 + c0 + kb + 4);
      f32x4 t2 = *(const f32x4*)(xrow1 + c0 + kb + 16);
      f32x4 t3 = *(const f32x4*)(xrow1 + c0 + kb + 20);
      a1[0]=(__bf16)t0[0]; a1[1]=(__bf16)t0[1]; a1[2]=(__bf16)t0[2]; a1[3]=(__bf16)t0[3];
      a1[4]=(__bf16)t1[0]; a1[5]=(__bf16)t1[1]; a1[6]=(__bf16)t1[2]; a1[7]=(__bf16)t1[3];
      a1[8]=(__bf16)t2[0]; a1[9]=(__bf16)t2[1]; a1[10]=(__bf16)t2[2]; a1[11]=(__bf16)t2[3];
      a1[12]=(__bf16)t3[0]; a1[13]=(__bf16)t3[1]; a1[14]=(__bf16)t3[2]; a1[15]=(__bf16)t3[3];
    }

    // ---- 8 N-tiles; each B fragment feeds both A fragments ----
#pragma unroll
    for (int nt = 0; nt < 8; ++nt) {
      const unsigned short* wp = w + (size_t)(nt * 16 + r) * DM + c0 + kb;
      V16U bt; bt.q[0] = *(const u32x4*)(wp); bt.q[1] = *(const u32x4*)(wp + 16);
      v16bf bfr = bt.v;
      acc0[nt] = __builtin_amdgcn_wmma_f32_16x16x32_bf16(
          false, a0, false, bfr, (short)0, acc0[nt], false, false);
      acc1[nt] = __builtin_amdgcn_wmma_f32_16x16x32_bf16(
          false, a1, false, bfr, (short)0, acc1[nt], false, false);
    }
  }

  // ---- store bf16: Q row-major, Kx row-major at offset NFIX, Vt transposed
#pragma unroll
  for (int qt = 0; qt < 2; ++qt) {
#pragma unroll
    for (int i = 0; i < 8; ++i) {
      int m = m0 + qt * 16 + half * 8 + i;          // global token
      int b = m >> 12;
      int t = m & 4095;
#pragma unroll
      for (int nt = 0; nt < 8; ++nt) {
        unsigned short bits = f2bf(qt ? acc1[nt][i] : acc0[nt][i]);
        int col = nt * 16 + r;
        if (mat == 0)      Q [(size_t)m * DK + col] = bits;
        else if (mat == 1) Kx[((size_t)b * SKV + NFIX + t) * DK + col] = bits;
        else               Vt[((size_t)b * DV + col) * SKV + NFIX + t] = bits;
      }
    }
  }
}

// ---------------------------------------------------------------------------
// Kernel 3: flash attention, one wave per 16 query rows.
// Per 32-key step: 8 score WMMAs, V fragments loaded into registers BEFORE
// the softmax (so the one LDS barrier doesn't serialize them), online softmax
// with shfl_xor butterflies, P staged via LDS, 8 PV WMMAs. Next-iteration K
// loads overlap the PV WMMAs; next chunk prefetched via global_prefetch.
// ---------------------------------------------------------------------------
__global__ __launch_bounds__(32) void attn_kernel(
    const unsigned short* __restrict__ Q, const unsigned short* __restrict__ Kx,
    const unsigned short* __restrict__ Vt, float* __restrict__ out) {
  __shared__ unsigned short Pl[16][40];             // 16 rows x 32 keys (+pad)

  const int lane = threadIdx.x;
  const int r    = lane & 15;
  const int half = lane >> 4;
  const int kb   = half * 8;
  const int b    = blockIdx.y;
  const int t0   = blockIdx.x * 16;

  // ---- Q as 4 A-fragments (contraction dk=128 = 4 chunks of 32) ----
  const unsigned short* qrow = Q + (size_t)(b * TT + t0 + r) * DK;
  v16bf qa[4];
#pragma unroll
  for (int kc = 0; kc < 4; ++kc) {
    V16U t;
    t.q[0] = *(const u32x4*)(qrow + kc * 32 + kb);
    t.q[1] = *(const u32x4*)(qrow + kc * 32 + 16 + kb);
    qa[kc] = t.v;
  }

  float mrun[8], lrun[8];
#pragma unroll
  for (int i = 0; i < 8; ++i) { mrun[i] = -3.0e38f; lrun[i] = 0.0f; }
  v8f o[8] = {};

  const unsigned short* kbase = Kx + (size_t)b * SKV * DK;
  const unsigned short* vbase = Vt + (size_t)b * DV * SKV;
  const float kscale = 0.08838834764831845f * 1.4426950408889634f; // dk^-0.5*log2e

  for (int s0 = 0; s0 < SKV; s0 += 32) {            // 130 iterations exactly
    // ---- scores: two 16x16 tiles for keys [s0..+15] and [s0+16..+31] ----
    v8f st0 = {}, st1 = {};
#pragma unroll
    for (int kc = 0; kc < 4; ++kc) {
      const unsigned short* kp0 = kbase + (size_t)(s0 + r)      * DK + kc * 32 + kb;
      const unsigned short* kp1 = kbase + (size_t)(s0 + 16 + r) * DK + kc * 32 + kb;
      V16U u0, u1;
      u0.q[0] = *(const u32x4*)(kp0); u0.q[1] = *(const u32x4*)(kp0 + 16);
      u1.q[0] = *(const u32x4*)(kp1); u1.q[1] = *(const u32x4*)(kp1 + 16);
      v16bf k0 = u0.v;
      v16bf k1 = u1.v;
      st0 = __builtin_amdgcn_wmma_f32_16x16x32_bf16(false, qa[kc], false, k0, (short)0, st0, false, false);
      st1 = __builtin_amdgcn_wmma_f32_16x16x32_bf16(false, qa[kc], false, k1, (short)0, st1, false, false);
    }

    // ---- V fragments into registers NOW: latency hides under softmax ----
    v16bf vf[8];
#pragma unroll
    for (int nt = 0; nt < 8; ++nt) {
      const unsigned short* vp = vbase + (size_t)(nt * 16 + r) * SKV + s0 + kb;
      V16U uv; uv.q[0] = *(const u32x4*)(vp); uv.q[1] = *(const u32x4*)(vp + 16);
      vf[nt] = uv.v;
    }

    // ---- prefetch next 32-key chunk of K and V into cache ----
    if (s0 + 32 < SKV) {
      __builtin_prefetch(kbase + (size_t)(s0 + 32 + lane) * DK, 0, 3);
      __builtin_prefetch(kbase + (size_t)(s0 + 32 + lane) * DK + 64, 0, 3);
#pragma unroll
      for (int j = 0; j < 4; ++j)
        __builtin_prefetch(vbase + (size_t)(j * 32 + lane) * SKV + s0 + 32, 0, 3);
    }

    // ---- online softmax (exp2 domain); row (lane>>4)*8+i lives across 16
    //      lanes -> xor masks 1,2,4,8 stay within each 16-lane half ----
    float p0[8], p1[8], alpha[8];
#pragma unroll
    for (int i = 0; i < 8; ++i) {
      float a0 = st0[i] * kscale, a1 = st1[i] * kscale;
      float mx = fmaxf(a0, a1);
      mx = fmaxf(mx, __shfl_xor(mx, 1, 32));
      mx = fmaxf(mx, __shfl_xor(mx, 2, 32));
      mx = fmaxf(mx, __shfl_xor(mx, 4, 32));
      mx = fmaxf(mx, __shfl_xor(mx, 8, 32));
      float mn = fmaxf(mrun[i], mx);
      alpha[i] = __builtin_amdgcn_exp2f(mrun[i] - mn);
      p0[i]    = __builtin_amdgcn_exp2f(a0 - mn);
      p1[i]    = __builtin_amdgcn_exp2f(a1 - mn);
      float rs = p0[i] + p1[i];
      rs += __shfl_xor(rs, 1, 32);
      rs += __shfl_xor(rs, 2, 32);
      rs += __shfl_xor(rs, 4, 32);
      rs += __shfl_xor(rs, 8, 32);
      lrun[i] = lrun[i] * alpha[i] + rs;
      mrun[i] = mn;
    }

    // ---- P tile (C/D layout) -> LDS (issue stores before the acc-rescale
    //      VALU so the DS pipe drains under it) ----
#pragma unroll
    for (int i = 0; i < 8; ++i) {
      Pl[half * 8 + i][r]      = f2bf(p0[i]);
      Pl[half * 8 + i][16 + r] = f2bf(p1[i]);
    }
#pragma unroll
    for (int nt = 0; nt < 8; ++nt)
#pragma unroll
      for (int i = 0; i < 8; ++i) o[nt][i] *= alpha[i];

    asm volatile("s_wait_dscnt 0" ::: "memory");    // cross-lane store->load
    V16U pf;
    pf.q[0] = *(const u32x4*)(&Pl[r][kb]);
    pf.q[1] = *(const u32x4*)(&Pl[r][16 + kb]);
    v16bf pfrag = pf.v;
    // (no second barrier: per-wave DS ops are in-order, and the compiler
    //  preserves store/load order on Pl, so next iter's stores cannot pass
    //  these loads)

    // ---- P @ V: contraction over the 32 keys of this step ----
#pragma unroll
    for (int nt = 0; nt < 8; ++nt)
      o[nt] = __builtin_amdgcn_wmma_f32_16x16x32_bf16(
          false, pfrag, false, vf[nt], (short)0, o[nt], false, false);
  }

  // ---- epilogue: normalize, store fp32 output [B,T,128] ----
  float* obase = out + (size_t)(b * TT + t0) * DV;
#pragma unroll
  for (int i = 0; i < 8; ++i) {
    float inv = 1.0f / lrun[i];
    int mr = half * 8 + i;
#pragma unroll
    for (int nt = 0; nt < 8; ++nt)
      obase[(size_t)mr * DV + nt * 16 + r] = o[nt][i] * inv;
  }
}

// ---------------------------------------------------------------------------
// Workspace layout (bytes):
//   [0)          Wb  bf16 [3][128][1024]   786432
//   [786432)     Q   bf16 [32768][128]    8388608
//   [9175040)    Kx  bf16 [8][4160][128]  8519680
//   [17694720)   Vt  bf16 [8][128][4160]  8519680   total 26214400 (~25 MiB)
// ---------------------------------------------------------------------------
extern "C" void kernel_launch(void* const* d_in, const int* in_sizes, int n_in,
                              void* d_out, int out_size, void* d_ws, size_t ws_size,
                              hipStream_t stream) {
  const float* x  = (const float*)d_in[0];
  const float* Wq = (const float*)d_in[1];
  const float* Wk = (const float*)d_in[2];
  const float* Wv = (const float*)d_in[3];
  const float* fk = (const float*)d_in[4];
  const float* fv = (const float*)d_in[5];
  float* out = (float*)d_out;

  char* ws = (char*)d_ws;
  unsigned short* Wb = (unsigned short*)(ws);
  unsigned short* Qw = (unsigned short*)(ws + 786432);
  unsigned short* Kx = (unsigned short*)(ws + 9175040);
  unsigned short* Vt = (unsigned short*)(ws + 17694720);

  prep_weights<<<dim3((3 * DK * DM) / 256), 256, 0, stream>>>(Wq, Wk, Wv, Wb);
  prep_fixed  <<<dim3((BB * NFIX * DK) / 256), 256, 0, stream>>>(fk, fv, Kx, Vt);
  qkv_proj    <<<dim3(MTOT / 32, 3), 32, 0, stream>>>(x, Wb, Qw, Kx, Vt);
  attn_kernel <<<dim3(TT / 16, BB), 32, 0, stream>>>(Qw, Kx, Vt, out);
}
